// RoPEAttentionBlock_30829275250974
// MI455X (gfx1250) — matrix operand order repaired and checked by reading, not compiled, not run on hardware
//
#include <hip/hip_runtime.h>

// ---------------------------------------------------------------------------
// RoPE attention block for MI455X (gfx1250), wave32 + WMMA f16->f32.
// B=2, L=2048, D=1024, H=16, Hd=64.
// GEMMs register-blocked 64x64 (QKV) / 32x64 (proj) per wave for L2-BW relief.
// ---------------------------------------------------------------------------

typedef __attribute__((ext_vector_type(16))) _Float16 v16h;
typedef __attribute__((ext_vector_type(8)))  _Float16 v8h;
typedef __attribute__((ext_vector_type(8)))  float    v8f;

static constexpr int BB  = 2;
static constexpr int LL  = 2048;
static constexpr int DD  = 1024;
static constexpr int ND3 = 3072;
static constexpr int HH  = 16;
static constexpr int HD  = 64;

__device__ __forceinline__ v8f wmma_f16(v16h a, v16h b, v8f c) {
  // D = A(16x32 f16) * B(32x16 f16) + C(16x16 f32)
  return __builtin_amdgcn_wmma_f32_16x16x32_f16(
      /*neg_a=*/false, a, /*neg_b=*/false, b,
      /*c_mod=*/(short)0, c, /*reuse_a=*/false, /*reuse_b=*/false);
}

// A fragment: 16x32 tile of row-major A (leading dim lda, element stride 1).
// Lane r (=lane&15) holds row r; s (=lane>>4) selects K-halves per ISA table:
//   v0..3 : K = 8s + {0..7}; v4..7 : K = 16 + 8s + {0..7}
__device__ __forceinline__ v16h frag_a(const _Float16* __restrict__ A, int lda, int lane) {
  const int r = lane & 15, s = lane >> 4;
  const _Float16* p = A + (size_t)r * lda + s * 8;
  v8h lo = *(const v8h*)(p);
  v8h hi = *(const v8h*)(p + 16);
  v16h out;
#pragma unroll
  for (int i = 0; i < 8; ++i) { out[i] = lo[i]; out[i + 8] = hi[i]; }
  return out;
}

// B fragment: 32x16 tile (K x N), stored N-major: Bt[n][k], leading dim ldb.
// Lane c (=lane&15) holds column n=c; v0..3 : K = 8s + {0..7}; v4..7 : K = 16 + 8s + {0..7}
__device__ __forceinline__ v16h frag_b(const _Float16* __restrict__ Bt, int ldb, int lane) {
  const int c = lane & 15, s = lane >> 4;
  const _Float16* p = Bt + (size_t)c * ldb + s * 8;
  v8h lo = *(const v8h*)(p);
  v8h hi = *(const v8h*)(p + 16);
  v16h out;
#pragma unroll
  for (int i = 0; i < 8; ++i) { out[i] = lo[i]; out[i + 8] = hi[i]; }
  return out;
}

// Register-blocked accumulation: MI x NI grid of 16x16 tiles, K = DD.
// A row-major [.. ][DD], Bt N-major [..][DD].
template <int MI, int NI>
__device__ __forceinline__ void gemm_block(const _Float16* __restrict__ A,
                                           const _Float16* __restrict__ Bt,
                                           v8f acc[MI][NI], int lane) {
  for (int k = 0; k < DD; k += 32) {
    if (k + 64 < DD) {
      __builtin_prefetch(A + k + 64, 0, 0);     // global_prefetch_b8
      __builtin_prefetch(Bt + k + 64, 0, 0);
    }
    v16h af[MI], bf[NI];
#pragma unroll
    for (int i = 0; i < MI; ++i) af[i] = frag_a(A + (size_t)(16 * i) * DD + k, DD, lane);
#pragma unroll
    for (int j = 0; j < NI; ++j) bf[j] = frag_b(Bt + (size_t)(16 * j) * DD + k, DD, lane);
#pragma unroll
    for (int i = 0; i < MI; ++i)
#pragma unroll
      for (int j = 0; j < NI; ++j)
        acc[i][j] = wmma_f16(af[i], bf[j], acc[i][j]);   // MI*NI independent chains
  }
}

// --------------------------- conversion kernels ----------------------------

__global__ void cvt_f32_f16(const float* __restrict__ src, _Float16* __restrict__ dst, int n) {
  int i = blockIdx.x * blockDim.x + threadIdx.x;
  if (i < n) dst[i] = (_Float16)src[i];
}

// src[rows][cols] f32 -> dst[cols][rows] f16 (N-major weights for B-fragments)
__global__ void transpose_f32_f16(const float* __restrict__ src, _Float16* __restrict__ dst,
                                  int rows, int cols) {
  int i = blockIdx.x * blockDim.x + threadIdx.x;
  if (i >= rows * cols) return;
  int c = i % cols, r = i / cols;
  dst[(size_t)c * rows + r] = (_Float16)src[(size_t)r * cols + c];
}

// ------------------------------- QKV GEMM ----------------------------------
// Raw[m][n] = sum_k Xh[m][k] * Wqkv[k][n] + bqkv[n], m = b*L + l (4096 rows).
// 64x64 tile per wave: 64 mt x 48 nt = 3072 waves.
__global__ void __launch_bounds__(128)
qkv_gemm(const _Float16* __restrict__ Xh, const _Float16* __restrict__ Wt,
         const float* __restrict__ bqkv, _Float16* __restrict__ Raw) {
  const int lane = threadIdx.x & 31;
  const int wave = threadIdx.x >> 5;
  int tile = blockIdx.x * 4 + wave;                 // 3072 tiles
  const int NT = ND3 / 64;                          // 48
  int nt = tile % NT;
  int mt = tile / NT;                               // 0..63
  const int m0 = mt * 64, n0 = nt * 64;

  v8f acc[4][4] = {};
  gemm_block<4, 4>(Xh + (size_t)m0 * DD, Wt + (size_t)n0 * DD, acc, lane);

  const int c = lane & 15, s = lane >> 4;
#pragma unroll
  for (int j = 0; j < 4; ++j) {
    const float bias = bqkv[n0 + 16 * j + c];
#pragma unroll
    for (int i = 0; i < 4; ++i)
#pragma unroll
      for (int v = 0; v < 8; ++v) {
        int row = m0 + 16 * i + v + 8 * s;          // D-layout: M = v + 8s, N = c
        Raw[(size_t)row * ND3 + n0 + 16 * j + c] = (_Float16)(acc[i][j][v] + bias);
      }
  }
}

// ----------------------------- RoPE + scatter ------------------------------
// Reference quirk: sin and cos are BOTH the concatenated [sin|cos] embedding:
//   out[d] = emb[d] * (x[d] + rotate_half(x)[d])
//   emb[d<32] = sin(l*theta_d), emb[d>=32] = cos(l*theta_{d-32})
//   rh[d<32] = -x[2d+1], rh[d>=32] = x[2(d-32)]
__global__ void rope_scatter(const _Float16* __restrict__ Raw,
                             _Float16* __restrict__ Q, _Float16* __restrict__ K,
                             _Float16* __restrict__ Vt) {
  int t = blockIdx.x * blockDim.x + threadIdx.x;    // B*H*L*HD = 4194304
  if (t >= BB * HH * LL * HD) return;
  int hd = t & 63;
  int l  = (t >> 6) & (LL - 1);
  int h  = (t >> 17) & (HH - 1);
  int b  = t >> 21;

  const _Float16* row  = Raw + ((size_t)b * LL + l) * ND3;
  const _Float16* qrow = row + h * HD;
  const _Float16* krow = row + DD + h * HD;

  float e, rhq, rhk;
  if (hd < 32) {
    float ang = (float)l * powf(10000.0f, -(float)hd / 32.0f);
    e   = sinf(ang);
    rhq = -(float)qrow[2 * hd + 1];
    rhk = -(float)krow[2 * hd + 1];
  } else {
    int j = hd - 32;
    float ang = (float)l * powf(10000.0f, -(float)j / 32.0f);
    e   = cosf(ang);
    rhq = (float)qrow[2 * j];
    rhk = (float)krow[2 * j];
  }
  size_t qidx = (((size_t)b * HH + h) * LL + l) * HD + hd;   // [B,H,L,Hd]
  Q[qidx] = (_Float16)(e * ((float)qrow[hd] + rhq));
  K[qidx] = (_Float16)(e * ((float)krow[hd] + rhk));
  size_t vidx = (((size_t)b * HH + h) * HD + hd) * LL + l;   // V^T: [B,H,Hd,L]
  Vt[vidx] = row[2 * DD + h * HD + hd];
}

// ------------------------- flash attention per wave ------------------------
// One wave owns 16 query rows of one (b,h); streams 32 keys per iteration.
__global__ void __launch_bounds__(128)
attn(const _Float16* __restrict__ Q, const _Float16* __restrict__ K,
     const _Float16* __restrict__ Vt, _Float16* __restrict__ Oh) {
  __shared__ _Float16 lds[4][16 * 32];
  const int lane = threadIdx.x & 31;
  const int wave = threadIdx.x >> 5;
  int wid = blockIdx.x * 4 + wave;                  // 4096 waves total
  int qt = wid & (LL / 16 - 1);
  int h  = (wid >> 7) & (HH - 1);
  int b  = wid >> 11;

  const _Float16* Qb = Q  + ((((size_t)b * HH + h) * LL) + qt * 16) * HD;
  const _Float16* Kb = K  + (((size_t)b * HH + h) * LL) * HD;
  const _Float16* Vb = Vt + (((size_t)b * HH + h) * HD) * LL;
  const int c = lane & 15, s = lane >> 4;

  const v16h aq0 = frag_a(Qb + 0,  HD, lane);       // hd 0..31
  const v16h aq1 = frag_a(Qb + 32, HD, lane);       // hd 32..63

  v8f o0 = {}, o1 = {}, o2 = {}, o3 = {};
  float m[8], lsum[8];
#pragma unroll
  for (int v = 0; v < 8; ++v) { m[v] = -1e30f; lsum[v] = 0.0f; }

  _Float16* P = lds[wave];

  for (int kb = 0; kb < LL; kb += 32) {
    // S = Q * K^T : B-fragment contraction dim = head dim (K row-major [L][64])
    v8f s0 = {}, s1 = {};
    s0 = wmma_f16(aq0, frag_b(Kb + (size_t)kb * HD + 0,  HD, lane), s0);
    s0 = wmma_f16(aq1, frag_b(Kb + (size_t)kb * HD + 32, HD, lane), s0);
    s1 = wmma_f16(aq0, frag_b(Kb + (size_t)(kb + 16) * HD + 0,  HD, lane), s1);
    s1 = wmma_f16(aq1, frag_b(Kb + (size_t)(kb + 16) * HD + 32, HD, lane), s1);

    __syncthreads();                                // P tile reusable
#pragma unroll
    for (int v = 0; v < 8; ++v) {
      float x0 = s0[v] * 0.125f;                    // 1/sqrt(64)
      float x1 = s1[v] * 0.125f;
      float mt = fmaxf(x0, x1);                     // row max across 16 lanes
      mt = fmaxf(mt, __shfl_xor(mt, 1));
      mt = fmaxf(mt, __shfl_xor(mt, 2));
      mt = fmaxf(mt, __shfl_xor(mt, 4));
      mt = fmaxf(mt, __shfl_xor(mt, 8));
      float mn = fmaxf(m[v], mt);
      float alpha = __expf(m[v] - mn);
      m[v] = mn;
      float p0 = __expf(x0 - mn);
      float p1 = __expf(x1 - mn);
      float rs = p0 + p1;
      rs += __shfl_xor(rs, 1);
      rs += __shfl_xor(rs, 2);
      rs += __shfl_xor(rs, 4);
      rs += __shfl_xor(rs, 8);
      lsum[v] = lsum[v] * alpha + rs;
      o0[v] *= alpha; o1[v] *= alpha; o2[v] *= alpha; o3[v] *= alpha;
      int row = v + 8 * s;                          // D-layout -> LDS row-major 16x32
      P[row * 32 + c]      = (_Float16)p0;
      P[row * 32 + 16 + c] = (_Float16)p1;
    }
    __syncthreads();

    // Reload P as an A-fragment (16 rows x 32 keys) from LDS
    v16h ap;
    {
      const _Float16* pp = P + (lane & 15) * 32 + s * 8;
#pragma unroll
      for (int i = 0; i < 8; ++i) { ap[i] = pp[i]; ap[i + 8] = pp[i + 16]; }
    }
    // O += P * V : Vt is [Hd][L] so B-fragment contraction (keys) is contiguous
    o0 = wmma_f16(ap, frag_b(Vb + (size_t)0  * LL + kb, LL, lane), o0);
    o1 = wmma_f16(ap, frag_b(Vb + (size_t)16 * LL + kb, LL, lane), o1);
    o2 = wmma_f16(ap, frag_b(Vb + (size_t)32 * LL + kb, LL, lane), o2);
    o3 = wmma_f16(ap, frag_b(Vb + (size_t)48 * LL + kb, LL, lane), o3);
  }

#pragma unroll
  for (int v = 0; v < 8; ++v) {
    int row = v + 8 * s;
    int l = qt * 16 + row;
    float inv = 1.0f / lsum[v];
    size_t base = ((size_t)b * LL + l) * DD + h * HD;     // Oh: [B,L,D]
    Oh[base + 0  + c] = (_Float16)(o0[v] * inv);
    Oh[base + 16 + c] = (_Float16)(o1[v] * inv);
    Oh[base + 32 + c] = (_Float16)(o2[v] * inv);
    Oh[base + 48 + c] = (_Float16)(o3[v] * inv);
  }
}

// --------------------------- output projection -----------------------------
// out[m][n] = sum_k Oh[m][k] * Wout[k][n] + bout[n].  32x64 tile per wave:
// 128 mt x 16 nt = 2048 waves.
__global__ void __launch_bounds__(128)
proj_gemm(const _Float16* __restrict__ Oh, const _Float16* __restrict__ Wt,
          const float* __restrict__ bout, float* __restrict__ out) {
  const int lane = threadIdx.x & 31;
  const int wave = threadIdx.x >> 5;
  int tile = blockIdx.x * 4 + wave;                 // 2048 tiles
  int nt = tile & 15;
  int mt = tile >> 4;                               // 0..127
  const int m0 = mt * 32, n0 = nt * 64;

  v8f acc[2][4] = {};
  gemm_block<2, 4>(Oh + (size_t)m0 * DD, Wt + (size_t)n0 * DD, acc, lane);

  const int c = lane & 15, s = lane >> 4;
#pragma unroll
  for (int j = 0; j < 4; ++j) {
    const float bias = bout[n0 + 16 * j + c];
#pragma unroll
    for (int i = 0; i < 2; ++i)
#pragma unroll
      for (int v = 0; v < 8; ++v) {
        int row = m0 + 16 * i + v + 8 * s;
        out[(size_t)row * DD + n0 + 16 * j + c] = acc[i][j][v] + bias;
      }
  }
}

// ------------------------------- launcher ----------------------------------

extern "C" void kernel_launch(void* const* d_in, const int* in_sizes, int n_in,
                              void* d_out, int out_size, void* d_ws, size_t ws_size,
                              hipStream_t stream) {
  (void)in_sizes; (void)n_in; (void)out_size; (void)ws_size;
  const float* x    = (const float*)d_in[0];
  const float* Wqkv = (const float*)d_in[1];
  const float* bqkv = (const float*)d_in[2];
  const float* Wout = (const float*)d_in[3];
  const float* bout = (const float*)d_in[4];
  float* out = (float*)d_out;

  char* ws = (char*)d_ws;
  // byte offsets (total 64 MB; Oh aliases the dead Raw region)
  _Float16* Xh   = (_Float16*)(ws + 0);            //  8 MB  x f16
  _Float16* Wqt  = (_Float16*)(ws + 8388608);      //  6 MB  Wqkv^T f16 [3072][1024]
  _Float16* Wot  = (_Float16*)(ws + 14680064);     //  2 MB  Wout^T f16 [1024][1024]
  _Float16* Raw  = (_Float16*)(ws + 16777216);     // 24 MB  raw qkv f16 [B][L][3072]
  _Float16* Qf   = (_Float16*)(ws + 41943040);     //  8 MB  [B,H,L,64]
  _Float16* Kf   = (_Float16*)(ws + 50331648);     //  8 MB  [B,H,L,64]
  _Float16* Vt   = (_Float16*)(ws + 58720256);     //  8 MB  [B,H,64,L]
  _Float16* Oh   = (_Float16*)(ws + 16777216);     //  8 MB  [B,L,D] (alias of Raw)

  const int nX = BB * LL * DD;                     // 4194304
  cvt_f32_f16<<<(nX + 255) / 256, 256, 0, stream>>>(x, Xh, nX);
  transpose_f32_f16<<<(DD * ND3 + 255) / 256, 256, 0, stream>>>(Wqkv, Wqt, DD, ND3);
  transpose_f32_f16<<<(DD * DD + 255) / 256, 256, 0, stream>>>(Wout, Wot, DD, DD);

  // QKV GEMM: 64 x 48 = 3072 tiles of 64x64, 4 waves/block
  qkv_gemm<<<3072 / 4, 128, 0, stream>>>(Xh, Wqt, bqkv, Raw);

  rope_scatter<<<(BB * HH * LL * HD + 255) / 256, 256, 0, stream>>>(Raw, Qf, Kf, Vt);

  // 2 * 16 * 128 = 4096 waves, 4 waves/block
  attn<<<4096 / 4, 128, 0, stream>>>(Qf, Kf, Vt, Oh);

  // projection: 128 x 16 = 2048 tiles of 32x64
  proj_gemm<<<2048 / 4, 128, 0, stream>>>(Oh, Wot, bout, out);
}
